// FinePreprocess_9405978378725
// MI455X (gfx1250) — matrix operand order repaired and checked by reading, not compiled, not run on hardware
//
#include <hip/hip_runtime.h>
#include <hip/hip_bf16.h>

typedef __bf16 bf16;
typedef __bf16 v8bf  __attribute__((ext_vector_type(8)));
typedef __bf16 v16bf __attribute__((ext_vector_type(16)));
typedef float  v8f   __attribute__((ext_vector_type(8)));
typedef unsigned int u32x4 __attribute__((ext_vector_type(4)));
typedef int          i32x4 __attribute__((ext_vector_type(4)));
typedef int          i32x8 __attribute__((ext_vector_type(8)));

#define B_   4
#define C0_  128
#define C1_  256
#define HF_  256
#define WF_  256
#define HC_  128
#define WC_  128
#define ROWP 40   // LDS row pitch in bf16: 32 payload + 8 pad = 80 bytes

// ---------------------------------------------------------------------------
// TDM: 2D tile -> LDS, bf16, LDS pad 16B per 64B row (matches ROWP layout).
// D# packing per CDNA5 ISA.
// ---------------------------------------------------------------------------
__device__ __forceinline__ void tdm_load_2d(unsigned lds_off, const void* gaddr,
                                            int dim0, int dim1, int tile0,
                                            int tile1, int stride0)
{
  unsigned long long ga = (unsigned long long)(uintptr_t)gaddr;
  u32x4 g0;
  g0[0] = 1u;                                   // count=1
  g0[1] = lds_off;                              // lds_addr
  g0[2] = (unsigned)(ga & 0xffffffffull);       // global_addr lo
  g0[3] = (unsigned)(ga >> 32) | (2u << 30);    // global_addr hi | type=2
  i32x8 g1;
  g1[0] = 0x06D10000;                           // 2B elems, pad_en, 16DW+4DW
  g1[1] = (dim0 & 0xffff) << 16;
  g1[2] = ((dim0 >> 16) & 0xffff) | ((dim1 & 0xffff) << 16);
  g1[3] = ((dim1 >> 16) & 0xffff) | (tile0 << 16);
  g1[4] = (tile1 & 0xffff);
  g1[5] = stride0;
  g1[6] = 0;
  g1[7] = 0;
  i32x4 z = {0, 0, 0, 0};
#if defined(__clang_major__) && __clang_major__ >= 23
  i32x8 z8 = {};
  __builtin_amdgcn_tensor_load_to_lds(g0, g1, z, z, z8, 0);
#else
  __builtin_amdgcn_tensor_load_to_lds(g0, g1, z, z, 0);
#endif
}

// Fragment loaders: each exactly 2x ds_load_b128 from padded LDS rows.
__device__ __forceinline__ v16bf frag_a(const bf16* row, int lhalf) {
  v8bf lo = *(const v8bf*)(row + 8 * lhalf);
  v8bf hi = *(const v8bf*)(row + 16 + 8 * lhalf);
  return __builtin_shufflevector(lo, hi, 0, 1, 2, 3, 4, 5, 6, 7, 8, 9, 10, 11,
                                 12, 13, 14, 15);
}
__device__ __forceinline__ v16bf frag_b(const bf16* row, int lhalf) {
  v8bf lo = *(const v8bf*)(row + 8 * lhalf);
  v8bf hi = *(const v8bf*)(row + 16 + 8 * lhalf);
  return __builtin_shufflevector(lo, hi, 0, 1, 2, 3, 4, 5, 6, 7, 8, 9, 10, 11,
                                 12, 13, 14, 15);
}

// ---------------------------------------------------------------------------
// Weight prep: f32 -> bf16; 3x3 weights transposed to [co][tap][ci].
// ---------------------------------------------------------------------------
__global__ __launch_bounds__(256)
void k_prep_cast(const float* __restrict__ w, bf16* __restrict__ o, int n)
{
  int i = blockIdx.x * 256 + threadIdx.x;
  if (i < n) o[i] = (bf16)w[i];
}

__global__ __launch_bounds__(256)
void k_prep_w3(const float* __restrict__ w, bf16* __restrict__ o, int n)
{
  int i = blockIdx.x * 256 + threadIdx.x;   // i = ((co*9+tap)*256 + ci)
  if (i >= n) return;
  int ci  = i & 255;
  int rst = i >> 8;
  int tap = rst % 9;
  int co  = rst / 9;
  o[i] = (bf16)w[((size_t)co * 256 + ci) * 9 + tap];
}

// ---------------------------------------------------------------------------
// Kernel 1: 1x1 conv on coarse -> f32 t.  Double-buffered K pipeline.
// ---------------------------------------------------------------------------
__global__ __launch_bounds__(256)
void k_proj_coarse(const float* __restrict__ c0, const float* __restrict__ c1,
                   const bf16* __restrict__ wgt, float* __restrict__ t)
{
  __shared__ bf16 As[2][32 * ROWP];
  __shared__ bf16 BsT[2][64 * ROWP];
  const int tid   = threadIdx.x;
  const int lane  = tid & 31;
  const int wave  = tid >> 5;
  const int lhalf = lane >> 4;
  const int lmod  = lane & 15;
  const int wco   = wave >> 2;
  const int wpx   = wave & 3;
  const int p0    = blockIdx.x * 64;
  const int co0   = blockIdx.y * 32;
  const int b     = blockIdx.z;
  const float* src = (b < B_) ? (c0 + (size_t)b * C1_ * (HC_ * WC_))
                              : (c1 + (size_t)(b - B_) * C1_ * (HC_ * WC_));

  auto stage = [&](int ci0, int bi) {
    if (wave == 0)
      tdm_load_2d((unsigned)(uintptr_t)&As[bi][0],
                  wgt + (size_t)co0 * C1_ + ci0, C1_, C1_, 32, 32, C1_);
#pragma unroll
    for (int k = 0; k < 2; ++k) {           // 512 quads / 256 threads
      int q  = tid + k * 256;
      int r  = q >> 4;                      // ci row 0..31
      int c4 = (q & 15) * 4;                // px 0..60 step 4
      float4 f = *(const float4*)(src + (size_t)(ci0 + r) * (HC_ * WC_) + p0 + c4);
      bf16* d = &BsT[bi][c4 * ROWP + r];
      d[0 * ROWP] = (bf16)f.x;
      d[1 * ROWP] = (bf16)f.y;
      d[2 * ROWP] = (bf16)f.z;
      d[3 * ROWP] = (bf16)f.w;
    }
  };

  v8f acc = {};
  stage(0, 0);
  if (wave == 0) __builtin_amdgcn_s_wait_tensorcnt(0);
  __syncthreads();
#pragma unroll 1
  for (int c = 0; c < C1_ / 32; ++c) {
    const int cur = c & 1;
    if (c + 1 < C1_ / 32) stage((c + 1) * 32, cur ^ 1);
    v16bf av = frag_a(&As[cur][(16 * wco + lmod) * ROWP], lhalf);
    v16bf bv = frag_b(&BsT[cur][(16 * wpx + lmod) * ROWP], lhalf);
    acc = __builtin_amdgcn_wmma_f32_16x16x32_bf16(false, av, false, bv,
                                                  (short)0, acc, false, false);
    if (wave == 0) __builtin_amdgcn_s_wait_tensorcnt(0);
    __syncthreads();
  }
  const int px = p0 + 16 * wpx + lmod;
#pragma unroll
  for (int r = 0; r < 8; ++r) {
    int co = co0 + 16 * wco + r + 8 * lhalf;
    t[((size_t)b * C1_ + co) * (HC_ * WC_) + px] = acc[r];
  }
}

// ---------------------------------------------------------------------------
// Kernel 2: 1x1 conv on fine + fused bilinear up2(t) -> bf16 x0.
// ---------------------------------------------------------------------------
__global__ __launch_bounds__(256)
void k_proj_fine_up(const float* __restrict__ f0, const float* __restrict__ f1,
                    const bf16* __restrict__ wgt, const float* __restrict__ t,
                    bf16* __restrict__ xo)
{
  __shared__ bf16 As[2][32 * ROWP];
  __shared__ bf16 BsT[2][64 * ROWP];
  const int tid   = threadIdx.x;
  const int lane  = tid & 31;
  const int wave  = tid >> 5;
  const int lhalf = lane >> 4;
  const int lmod  = lane & 15;
  const int wco   = wave >> 2;
  const int wpx   = wave & 3;
  const int bx    = blockIdx.x;
  const int w0    = (bx & 3) * 64;
  const int h     = bx >> 2;
  const int co0   = blockIdx.y * 32;
  const int b     = blockIdx.z;
  const float* src = (b < B_) ? (f0 + (size_t)b * C0_ * (HF_ * WF_))
                              : (f1 + (size_t)(b - B_) * C0_ * (HF_ * WF_));
  const float* srcRow = src + (size_t)h * WF_ + w0;

  auto stage = [&](int ci0, int bi) {
    if (wave == 0)
      tdm_load_2d((unsigned)(uintptr_t)&As[bi][0],
                  wgt + (size_t)co0 * C0_ + ci0, C0_, C1_, 32, 32, C0_);
#pragma unroll
    for (int k = 0; k < 2; ++k) {
      int q  = tid + k * 256;
      int r  = q >> 4;
      int c4 = (q & 15) * 4;
      float4 f = *(const float4*)(srcRow + (size_t)(ci0 + r) * (HF_ * WF_) + c4);
      bf16* d = &BsT[bi][c4 * ROWP + r];
      d[0 * ROWP] = (bf16)f.x;
      d[1 * ROWP] = (bf16)f.y;
      d[2 * ROWP] = (bf16)f.z;
      d[3 * ROWP] = (bf16)f.w;
    }
  };

  v8f acc = {};
  stage(0, 0);
  if (wave == 0) __builtin_amdgcn_s_wait_tensorcnt(0);
  __syncthreads();
#pragma unroll 1
  for (int c = 0; c < C0_ / 32; ++c) {
    const int cur = c & 1;
    if (c + 1 < C0_ / 32) stage((c + 1) * 32, cur ^ 1);
    v16bf av = frag_a(&As[cur][(16 * wco + lmod) * ROWP], lhalf);
    v16bf bv = frag_b(&BsT[cur][(16 * wpx + lmod) * ROWP], lhalf);
    acc = __builtin_amdgcn_wmma_f32_16x16x32_bf16(false, av, false, bv,
                                                  (short)0, acc, false, false);
    if (wave == 0) __builtin_amdgcn_s_wait_tensorcnt(0);
    __syncthreads();
  }
  const float s = 127.0f / 255.0f;
  float posH = h * s;
  int   i0   = (int)posH; if (i0 > 126) i0 = 126;
  float fh   = posH - (float)i0;
  const int wg = w0 + 16 * wpx + lmod;
  float posW = wg * s;
  int   j0   = (int)posW; if (j0 > 126) j0 = 126;
  float fw   = posW - (float)j0;
#pragma unroll
  for (int r = 0; r < 8; ++r) {
    int co = co0 + 16 * wco + r + 8 * lhalf;
    const float* tp = t + ((size_t)b * C1_ + co) * (HC_ * WC_);
    float v00 = tp[(size_t)i0 * WC_ + j0];
    float v01 = tp[(size_t)i0 * WC_ + j0 + 1];
    float v10 = tp[(size_t)(i0 + 1) * WC_ + j0];
    float v11 = tp[(size_t)(i0 + 1) * WC_ + j0 + 1];
    float up = (v00 * (1.f - fw) + v01 * fw) * (1.f - fh)
             + (v10 * (1.f - fw) + v11 * fw) * fh;
    xo[((size_t)b * C1_ + co) * (HF_ * WF_) + (size_t)h * WF_ + wg] =
        (bf16)(acc[r] + up);
  }
}

// ---------------------------------------------------------------------------
// Kernels 3/4: implicit-GEMM 3x3 conv, double-buffered.  Weights via TDM
// ([co][tap][ci] rows); input tile transposed to [hr][wl][ci] with an
// 8-aligned halo window (wl 0..79 maps to gw = w0-8+wl) for aligned v8bf loads.
// ---------------------------------------------------------------------------
template <int CO, int CI, bool BNACT>
__global__ __launch_bounds__(256)
void k_conv3x3(const bf16* __restrict__ in, const bf16* __restrict__ wT,
               const float* __restrict__ g, const float* __restrict__ bet,
               const float* __restrict__ mu, const float* __restrict__ var,
               bf16* __restrict__ out_bf, float* __restrict__ out_f)
{
  __shared__ bf16 WsT[2][288 * ROWP];       // rows: co_l*9 + tap
  __shared__ bf16 IsT[2][240 * ROWP];       // rows: hr*80 + wl, cols ci
  const int tid   = threadIdx.x;
  const int lane  = tid & 31;
  const int wave  = tid >> 5;
  const int lhalf = lane >> 4;
  const int lmod  = lane & 15;
  const int wco   = wave >> 2;
  const int wpx   = wave & 3;
  const int bx    = blockIdx.x;
  const int w0    = (bx & 3) * 64;
  const int h     = bx >> 2;
  const int co0   = blockIdx.y * 32;
  const int b     = blockIdx.z;
  const bf16* inb = in + (size_t)b * CI * (HF_ * WF_);

  auto stage = [&](int ci0, int bi) {
    if (wave == 0)
      tdm_load_2d((unsigned)(uintptr_t)&WsT[bi][0],
                  wT + ((size_t)co0 * 9) * CI + ci0, CI, CO * 9, 32, 288, CI);
    // 32ci x 3h x 10 chunks-of-8w = 960 vector jobs
#pragma unroll
    for (int k = 0; k < 4; ++k) {
      int j = tid + k * 256;
      if (j < 960) {
        int cil = j / 30;
        int rem = j % 30;
        int hr  = rem / 10;
        int wc  = rem % 10;
        int gh  = h + hr - 1;
        int gwb = w0 - 8 + wc * 8;
        const bf16* sp = inb + ((size_t)(ci0 + cil) * HF_ + gh) * WF_ + gwb;
        v8bf vv = {};
        if ((unsigned)gh < (unsigned)HF_) {
          if (gwb >= 0 && gwb + 8 <= WF_) {
            vv = *(const v8bf*)sp;            // aligned global_load_b128
          } else {
#pragma unroll
            for (int e = 0; e < 8; ++e) {
              int gw = gwb + e;
              if ((unsigned)gw < (unsigned)WF_)
                vv[e] = inb[((size_t)(ci0 + cil) * HF_ + gh) * WF_ + gw];
            }
          }
        }
        bf16* d = &IsT[bi][(hr * 80 + wc * 8) * ROWP + cil];
#pragma unroll
        for (int e = 0; e < 8; ++e) d[e * ROWP] = vv[e];
      }
    }
    if (ci0 + 32 < CI)  // global_prefetch_b8 hint for the chunk after next
      __builtin_prefetch(&inb[((size_t)(ci0 + 32) * HF_ + h) * WF_ + w0], 0, 0);
  };

  v8f acc = {};
  stage(0, 0);
  if (wave == 0) __builtin_amdgcn_s_wait_tensorcnt(0);
  __syncthreads();
  const int NC = CI / 32;
#pragma unroll 1
  for (int c = 0; c < NC; ++c) {
    const int cur = c & 1;
    if (c + 1 < NC) stage((c + 1) * 32, cur ^ 1);
#pragma unroll
    for (int tap = 0; tap < 9; ++tap) {
      const int dy = tap / 3, dx = tap % 3;
      v16bf av = frag_a(&WsT[cur][((16 * wco + lmod) * 9 + tap) * ROWP], lhalf);
      v16bf bv =
          frag_b(&IsT[cur][(dy * 80 + 16 * wpx + lmod + dx + 7) * ROWP], lhalf);
      acc = __builtin_amdgcn_wmma_f32_16x16x32_bf16(false, av, false, bv,
                                                    (short)0, acc, false, false);
    }
    if (wave == 0) __builtin_amdgcn_s_wait_tensorcnt(0);
    __syncthreads();
  }
  const int wg = w0 + 16 * wpx + lmod;
#pragma unroll
  for (int r = 0; r < 8; ++r) {
    int co = co0 + 16 * wco + r + 8 * lhalf;
    float y = acc[r];
    if (BNACT) {
      float inv = rsqrtf(var[co] + 1e-5f);
      y = (y - mu[co]) * (g[co] * inv) + bet[co];
      y = (y >= 0.f) ? y : 0.01f * y;
      out_bf[((size_t)b * CO + co) * (HF_ * WF_) + (size_t)h * WF_ + wg] = (bf16)y;
    } else {
      out_f[((size_t)b * CO + co) * (HF_ * WF_) + (size_t)h * WF_ + wg] = y;
    }
  }
}

// ---------------------------------------------------------------------------
// Crop gathers
// ---------------------------------------------------------------------------
__global__ __launch_bounds__(256)
void k_crop0(const float* __restrict__ y, const int* __restrict__ bidx,
             const int* __restrict__ iidx, float* __restrict__ out)
{
  size_t idx = (size_t)blockIdx.x * 256 + threadIdx.x;  // 2048*128*64 exact
  int cc = (int)(idx & 7);
  int r  = (int)((idx >> 3) & 7);
  int c  = (int)((idx >> 6) & 127);
  int n  = (int)(idx >> 13);
  int b  = bidx[n];
  int wi = iidx[n];
  int r0 = (wi >> 5) * 8;
  int c0 = (wi & 31) * 8;
  out[idx] = y[((size_t)b * C0_ + c) * (HF_ * WF_) +
               (size_t)(r0 + r) * WF_ + (c0 + cc)];
}

__global__ __launch_bounds__(256)
void k_crop1(const float* __restrict__ y, const int* __restrict__ bidx,
             const int* __restrict__ jidx, float* __restrict__ out)
{
  size_t idx = (size_t)blockIdx.x * 256 + threadIdx.x;  // 2048*128*100 exact
  if (idx >= (size_t)2048 * 128 * 100) return;
  int cc = (int)(idx % 10);
  int r  = (int)((idx / 10) % 10);
  int c  = (int)((idx / 100) % 128);
  int n  = (int)(idx / 12800);
  int b  = B_ + bidx[n];
  int wi = jidx[n];
  int gy = (wi >> 5) * 8 + r - 1;   // pad = 1
  int gx = (wi & 31) * 8 + cc - 1;
  float v = 0.f;
  if ((unsigned)gy < (unsigned)HF_ && (unsigned)gx < (unsigned)WF_)
    v = y[((size_t)b * C0_ + c) * (HF_ * WF_) + (size_t)gy * WF_ + gx];
  out[idx] = v;
}

// ---------------------------------------------------------------------------
extern "C" void kernel_launch(void* const* d_in, const int* in_sizes, int n_in,
                              void* d_out, int out_size, void* d_ws, size_t ws_size,
                              hipStream_t stream) {
  const float* x0f = (const float*)d_in[0];
  const float* x0c = (const float*)d_in[1];
  const float* x1f = (const float*)d_in[2];
  const float* x1c = (const float*)d_in[3];
  const float* p0w = (const float*)d_in[4];
  const float* plw = (const float*)d_in[5];
  const float* fw1 = (const float*)d_in[6];
  const float* bg  = (const float*)d_in[7];
  const float* bb  = (const float*)d_in[8];
  const float* bm  = (const float*)d_in[9];
  const float* bv  = (const float*)d_in[10];
  const float* fw2 = (const float*)d_in[11];
  const int* bidx  = (const int*)d_in[12];
  const int* iidx  = (const int*)d_in[13];
  const int* jidx  = (const int*)d_in[14];

  char* ws = (char*)d_ws;
  float* t     = (float*)(ws);                                    // 128 MB
  bf16*  x0buf = (bf16*)(ws + 134217728ull);                      // 256 MB
  bf16*  ybuf  = (bf16*)(ws + 402653184ull);                      // 256 MB
  float* obuf  = (float*)(ws + 671088640ull);                     // 256 MB
  bf16*  plwB  = (bf16*)(ws + 939524096ull);                      // 128 KB
  bf16*  p0wB  = (bf16*)(ws + 939524096ull + 131072ull);          // 64 KB
  bf16*  w1T   = (bf16*)(ws + 939524096ull + 196608ull);          // 1.13 MB
  bf16*  w2T   = (bf16*)(ws + 939524096ull + 1376256ull);         // 576 KB
  float* out   = (float*)d_out;

  k_prep_cast<<<256, 256, 0, stream>>>(plw, plwB, 65536);
  k_prep_cast<<<128, 256, 0, stream>>>(p0w, p0wB, 32768);
  k_prep_w3<<<2304, 256, 0, stream>>>(fw1, w1T, 589824);
  k_prep_w3<<<1152, 256, 0, stream>>>(fw2, w2T, 294912);

  k_proj_coarse<<<dim3(256, 8, 8), 256, 0, stream>>>(x0c, x1c, plwB, t);
  k_proj_fine_up<<<dim3(1024, 8, 8), 256, 0, stream>>>(x0f, x1f, p0wB, t, x0buf);
  k_conv3x3<C1_, C1_, true><<<dim3(1024, 8, 8), 256, 0, stream>>>(
      x0buf, w1T, bg, bb, bm, bv, ybuf, nullptr);
  k_conv3x3<C0_, C1_, false><<<dim3(1024, 4, 8), 256, 0, stream>>>(
      ybuf, w2T, nullptr, nullptr, nullptr, nullptr, nullptr, obuf);
  k_crop0<<<65536, 256, 0, stream>>>(obuf, bidx, iidx, out);
  k_crop1<<<102400, 256, 0, stream>>>(obuf, bidx, jidx, out + 16777216);
}
